// FeedForwardQuantum_65481071398824
// MI455X (gfx1250) — compile-verified, matrix operand order checked by reading, not run on hardware
//
#include <hip/hip_runtime.h>
#include <hip/hip_bf16.h>

// ---------------------------------------------------------------------------
// FeedForwardQuantum for MI455X (gfx1250)
//
//   ev  = closed-form quantum expvals (product state + CNOT basis permutation
//         -> prefix products of per-wire Bloch-z)
//   H   = relu(ev @ W1^T + b1) -> f16 workspace (128MB, L2-resident: 192MB L2)
//   out = H @ W2^T + b2        -> v_wmma_f32_16x16x32_f16, 64x64 tile per wave,
//         B staged through LDS with gfx1250 async-copy (double buffered)
// ---------------------------------------------------------------------------

typedef _Float16 half_t;
typedef half_t v16h __attribute__((ext_vector_type(16)));
typedef half_t v8h  __attribute__((ext_vector_type(8)));
typedef float  v8f  __attribute__((ext_vector_type(8)));

#define N_WIRES 8
#define FFN     2048
#define EMB     512
#define BPAD    40   // LDS column stride in halfs (80B) -> conflict-minimal ds_load_b128

#if defined(__AMDGCN__) && __has_builtin(__builtin_amdgcn_global_load_async_to_lds_b128)
#define USE_ASYNC_B 1
// Param types per hipcc diagnostic: pointer to 16B int vector, global (AS1)
// source / LDS (AS3) destination.
typedef int v4i __attribute__((vector_size(16)));
typedef __attribute__((address_space(1))) v4i* glb_v4i_p;
typedef __attribute__((address_space(3))) v4i* lds_v4i_p;
#if __has_builtin(__builtin_amdgcn_s_wait_asynccnt)
#define WAIT_ASYNC(n) __builtin_amdgcn_s_wait_asynccnt(n)
#else
#define WAIT_ASYNC(n) asm volatile("s_wait_asynccnt %0" ::"i"(n))
#endif
#else
#define USE_ASYNC_B 0
#endif

// ---------------------------------------------------------------------------
// Kernel 1: ev (closed form) + GEMM1 (K=8) + ReLU -> H_f16 [N, 2048]
// ---------------------------------------------------------------------------
__global__ void __launch_bounds__(256)
evh_kernel(const float* __restrict__ x,      // [N, 8]
           const float* __restrict__ params, // [8, 3]
           const float* __restrict__ W1,     // [2048, 8]
           const float* __restrict__ b1,     // [2048]
           half_t* __restrict__ Hf)          // [N, 2048] f16 out
{
    __shared__ float ev_s[64][N_WIRES];

    const int t = threadIdx.x;
    const int rowBase = blockIdx.x * 64;

    if (t < 64) {
        const int n = rowBase + t;
        float z[N_WIRES];
#pragma unroll
        for (int i = 0; i < N_WIRES; ++i) {
            const float a = params[i * 3 + 0];
            const float b = params[i * 3 + 1];
            const float c = params[i * 3 + 2];
            const float sa = __sinf(a), ca = __cosf(a);
            const float sb = __sinf(b), cb = __cosf(b);
            const float sc = __sinf(c), cc = __cosf(c);
            const float Ai = sa * sb * sc + ca * cc;   // coeff of cos(x)
            const float Bi = -cb * sc;                 // coeff of sin(x)
            const float xv = x[(size_t)n * N_WIRES + i];
            z[i] = Ai * __cosf(xv) + Bi * __sinf(xv);
        }
        // <Z_0> = z1..z7 ; <Z_k> = z0..zk (k>=1)  (CNOT chain = XOR basis perm)
        float p17 = z[1];
#pragma unroll
        for (int i = 2; i < N_WIRES; ++i) p17 *= z[i];
        ev_s[t][0] = p17;
        float pp = z[0];
#pragma unroll
        for (int k = 1; k < N_WIRES; ++k) { pp *= z[k]; ev_s[t][k] = pp; }
    }
    __syncthreads();

#pragma unroll
    for (int j = 0; j < FFN / 256; ++j) {
        const int f = j * 256 + t;
        float w[N_WIRES];
#pragma unroll
        for (int i = 0; i < N_WIRES; ++i) w[i] = W1[(size_t)f * N_WIRES + i];
        const float bias = b1[f];
        for (int r = 0; r < 64; ++r) {
            float acc = bias;
#pragma unroll
            for (int i = 0; i < N_WIRES; ++i) acc += ev_s[r][i] * w[i];
            acc = fmaxf(acc, 0.0f);
            Hf[(size_t)(rowBase + r) * FFN + f] = (half_t)acc;
        }
    }
}

// ---------------------------------------------------------------------------
// Kernel 2: W2 f32 -> f16 (row-major kept: W2 rows contiguous in K)
// ---------------------------------------------------------------------------
__global__ void __launch_bounds__(256)
cvt_w2_kernel(const float* __restrict__ W2, half_t* __restrict__ W2h, int n)
{
    const int i = blockIdx.x * 256 + threadIdx.x;
    if (i < n) W2h[i] = (half_t)W2[i];
}

// ---------------------------------------------------------------------------
// Fragment loads per ISA 16-bit A layout:
//   lane L: row/col = base + (L & 15), halfSel = L >> 4
//   elements 0..7  -> K = k0 +      halfSel*8 .. +7   (16B contiguous)
//   elements 8..15 -> K = k0 + 16 + halfSel*8 .. +7   (16B contiguous)
// ---------------------------------------------------------------------------
__device__ __forceinline__ v16h load_frag_g(const half_t* __restrict__ row,
                                            int k0, int halfSel)
{
    const v8h lo = *reinterpret_cast<const v8h*>(row + k0 + halfSel * 8);
    const v8h hi = *reinterpret_cast<const v8h*>(row + k0 + 16 + halfSel * 8);
    return __builtin_shufflevector(lo, hi, 0, 1, 2, 3, 4, 5, 6, 7,
                                           8, 9, 10, 11, 12, 13, 14, 15);
}

__device__ __forceinline__ v16h load_frag_l(const half_t* col, int halfSel)
{
    const v8h lo = *reinterpret_cast<const v8h*>(col + halfSel * 8);
    const v8h hi = *reinterpret_cast<const v8h*>(col + 16 + halfSel * 8);
    return __builtin_shufflevector(lo, hi, 0, 1, 2, 3, 4, 5, 6, 7,
                                           8, 9, 10, 11, 12, 13, 14, 15);
}

#define WMMA(A, B, C) \
    __builtin_amdgcn_wmma_f32_16x16x32_f16(false, A, false, B, (short)0, C, false, false)

// ---------------------------------------------------------------------------
// Kernel 3: out[N,512] = H[N,2048](f16) @ W2^T(f16) + b2, f32 accumulate.
// 256 threads = 8 waves; wave tile 64x64 (4x4 WMMA frags, 16 WMMA/k-step).
// Block tile 512x64. B (shared operand) staged in LDS via async copy.
// ---------------------------------------------------------------------------
__global__ void __launch_bounds__(256)
gemm2_wmma_kernel(const half_t* __restrict__ Hf,   // [N, 2048]
                  const half_t* __restrict__ W2h,  // [512, 2048]
                  const float*  __restrict__ b2,   // [512]
                  float* __restrict__ out)         // [N, 512]
{
    const int t    = threadIdx.x;
    const int wave = t >> 5;
    const int lane = t & 31;
    const int laneM   = lane & 15;
    const int halfSel = lane >> 4;

    const int m0 = blockIdx.x * 512 + wave * 64;  // 64 rows per wave
    const int n0 = blockIdx.y * 64;               // 64 cols per block

    const half_t* arow[4];
#pragma unroll
    for (int rt = 0; rt < 4; ++rt)
        arow[rt] = Hf + (size_t)(m0 + rt * 16 + laneM) * FFN;

    v8f acc[4][4] = {};

#if USE_ASYNC_B
    // LDS B tile: col n -> 32 K-halfs at stride BPAD (padded, 16B aligned)
    __shared__ __attribute__((aligned(16))) half_t Bs[2][64 * BPAD];
    const int tn = t >> 2;          // 0..63: column this thread copies
    const int tk = (t & 3) * 8;     // 0,8,16,24: K sub-chunk (16B)
    const half_t* gB = W2h + (size_t)(n0 + tn) * FFN + tk;
    half_t* lB0 = &Bs[0][tn * BPAD + tk];
    half_t* lB1 = &Bs[1][tn * BPAD + tk];

    // stage 0 in flight (1 async op per wave covers 512B; 8 waves -> 4KB tile)
    __builtin_amdgcn_global_load_async_to_lds_b128((glb_v4i_p)gB, (lds_v4i_p)lB0, 0, 0);

    for (int s = 0; s < FFN / 32; ++s) {
        const int k0 = s * 32;
        if (s + 1 < FFN / 32) {   // issue next stage, then wait for current
            __builtin_amdgcn_global_load_async_to_lds_b128(
                (glb_v4i_p)(gB + k0 + 32),
                (lds_v4i_p)(((s + 1) & 1) ? lB1 : lB0), 0, 0);
            WAIT_ASYNC(1);
        } else {
            WAIT_ASYNC(0);
        }
        __syncthreads();          // async LDS writes visible to all waves

        // speculative prefetch of A two k-steps ahead (safe past end)
        __builtin_prefetch(arow[0] + k0 + 64, 0, 3);
        __builtin_prefetch(arow[1] + k0 + 64, 0, 3);
        __builtin_prefetch(arow[2] + k0 + 64, 0, 3);
        __builtin_prefetch(arow[3] + k0 + 64, 0, 3);

        v16h a0 = load_frag_g(arow[0], k0, halfSel);
        v16h a1 = load_frag_g(arow[1], k0, halfSel);
        v16h a2 = load_frag_g(arow[2], k0, halfSel);
        v16h a3 = load_frag_g(arow[3], k0, halfSel);

        const half_t* buf = &Bs[s & 1][0];
        v16h bf0 = load_frag_l(buf + (0 * 16 + laneM) * BPAD, halfSel);
        v16h bf1 = load_frag_l(buf + (1 * 16 + laneM) * BPAD, halfSel);
        v16h bf2 = load_frag_l(buf + (2 * 16 + laneM) * BPAD, halfSel);
        v16h bf3 = load_frag_l(buf + (3 * 16 + laneM) * BPAD, halfSel);

        acc[0][0] = WMMA(a0, bf0, acc[0][0]);
        acc[1][0] = WMMA(a1, bf0, acc[1][0]);
        acc[2][0] = WMMA(a2, bf0, acc[2][0]);
        acc[3][0] = WMMA(a3, bf0, acc[3][0]);
        acc[0][1] = WMMA(a0, bf1, acc[0][1]);
        acc[1][1] = WMMA(a1, bf1, acc[1][1]);
        acc[2][1] = WMMA(a2, bf1, acc[2][1]);
        acc[3][1] = WMMA(a3, bf1, acc[3][1]);
        acc[0][2] = WMMA(a0, bf2, acc[0][2]);
        acc[1][2] = WMMA(a1, bf2, acc[1][2]);
        acc[2][2] = WMMA(a2, bf2, acc[2][2]);
        acc[3][2] = WMMA(a3, bf2, acc[3][2]);
        acc[0][3] = WMMA(a0, bf3, acc[0][3]);
        acc[1][3] = WMMA(a1, bf3, acc[1][3]);
        acc[2][3] = WMMA(a2, bf3, acc[2][3]);
        acc[3][3] = WMMA(a3, bf3, acc[3][3]);

        __syncthreads();          // reads done before buffer is overwritten
    }
#else
    // Fallback: direct global B loads (cached in L0/L2; W2f16 is only 2MB)
    const half_t* brow[4];
#pragma unroll
    for (int bt = 0; bt < 4; ++bt)
        brow[bt] = W2h + (size_t)(n0 + bt * 16 + laneM) * FFN;

    for (int k0 = 0; k0 < FFN; k0 += 32) {
        __builtin_prefetch(arow[0] + k0 + 64, 0, 3);
        __builtin_prefetch(arow[1] + k0 + 64, 0, 3);
        __builtin_prefetch(arow[2] + k0 + 64, 0, 3);
        __builtin_prefetch(arow[3] + k0 + 64, 0, 3);

        v16h a0 = load_frag_g(arow[0], k0, halfSel);
        v16h a1 = load_frag_g(arow[1], k0, halfSel);
        v16h a2 = load_frag_g(arow[2], k0, halfSel);
        v16h a3 = load_frag_g(arow[3], k0, halfSel);
        v16h bf0 = load_frag_g(brow[0], k0, halfSel);
        v16h bf1 = load_frag_g(brow[1], k0, halfSel);
        v16h bf2 = load_frag_g(brow[2], k0, halfSel);
        v16h bf3 = load_frag_g(brow[3], k0, halfSel);

        acc[0][0] = WMMA(a0, bf0, acc[0][0]);
        acc[1][0] = WMMA(a1, bf0, acc[1][0]);
        acc[2][0] = WMMA(a2, bf0, acc[2][0]);
        acc[3][0] = WMMA(a3, bf0, acc[3][0]);
        acc[0][1] = WMMA(a0, bf1, acc[0][1]);
        acc[1][1] = WMMA(a1, bf1, acc[1][1]);
        acc[2][1] = WMMA(a2, bf1, acc[2][1]);
        acc[3][1] = WMMA(a3, bf1, acc[3][1]);
        acc[0][2] = WMMA(a0, bf2, acc[0][2]);
        acc[1][2] = WMMA(a1, bf2, acc[1][2]);
        acc[2][2] = WMMA(a2, bf2, acc[2][2]);
        acc[3][2] = WMMA(a3, bf2, acc[3][2]);
        acc[0][3] = WMMA(a0, bf3, acc[0][3]);
        acc[1][3] = WMMA(a1, bf3, acc[1][3]);
        acc[2][3] = WMMA(a2, bf3, acc[2][3]);
        acc[3][3] = WMMA(a3, bf3, acc[3][3]);
    }
#endif

    // Epilogue: D VGPR v holds row M = v + 8*halfSel, col n = laneM (per tile).
#pragma unroll
    for (int bt = 0; bt < 4; ++bt) {
        const int n = n0 + bt * 16 + laneM;
        const float bias = b2[n];
#pragma unroll
        for (int rt = 0; rt < 4; ++rt) {
#pragma unroll
            for (int v = 0; v < 8; ++v) {
                const int M = m0 + rt * 16 + v + 8 * halfSel;
                out[(size_t)M * EMB + n] = acc[rt][bt][v] + bias;
            }
        }
    }
}

// ---------------------------------------------------------------------------
extern "C" void kernel_launch(void* const* d_in, const int* in_sizes, int n_in,
                              void* d_out, int out_size, void* d_ws, size_t ws_size,
                              hipStream_t stream)
{
    const float* x      = (const float*)d_in[0];  // [B,S,8]
    const float* params = (const float*)d_in[1];  // [8,3]
    const float* W1     = (const float*)d_in[2];  // [2048,8]
    const float* b1     = (const float*)d_in[3];  // [2048]
    const float* W2     = (const float*)d_in[4];  // [512,2048]
    const float* b2     = (const float*)d_in[5];  // [512]
    float* out = (float*)d_out;

    const int N = in_sizes[0] / N_WIRES;          // 32768 samples

    // Workspace: H_f16 [N,2048] then W2_f16 [512,2048]
    half_t* Hf  = (half_t*)d_ws;
    half_t* W2h = (half_t*)((char*)d_ws + (size_t)N * FFN * sizeof(half_t));

    const int w2n = EMB * FFN;
    cvt_w2_kernel<<<(w2n + 255) / 256, 256, 0, stream>>>(W2, W2h, w2n);

    evh_kernel<<<N / 64, 256, 0, stream>>>(x, params, W1, b1, Hf);

    dim3 grid(N / 512, EMB / 64);
    gemm2_wmma_kernel<<<grid, 256, 0, stream>>>(Hf, W2h, b2, out);
}
